// RNA_ATAC_Pairing_68307159876172
// MI455X (gfx1250) — matrix operand order repaired and checked by reading, not compiled, not run on hardware
//
#include <hip/hip_runtime.h>
#include <cstdint>
#include <cstddef>

// ---------------------------------------------------------------------------
// RNA-ATAC pairing model for MI455X (gfx1250, wave32, WMMA).
// All heavy GEMMs use V_WMMA_F32_16X16X32_BF16. Attention (QK^T, sigmoid,
// top-10, softmax, threshold) is fully fused — the 256MB score tensor is
// never materialized.
// ---------------------------------------------------------------------------

using bf16 = __bf16;
typedef bf16  v8bf  __attribute__((ext_vector_type(8)));
typedef bf16  v16bf __attribute__((ext_vector_type(16)));
typedef float v8f   __attribute__((ext_vector_type(8)));

#define NR    2048
#define NA    8192
#define NHEAD 4
#define TOPK  10

__device__ __forceinline__ v16bf cat16(v8bf lo, v8bf hi) {
  return __builtin_shufflevector(lo, hi, 0,1,2,3,4,5,6,7,8,9,10,11,12,13,14,15);
}
__device__ __forceinline__ v8f wmma_bf16(v16bf a, v16bf b, v8f c) {
  return __builtin_amdgcn_wmma_f32_16x16x32_bf16(false, a, false, b, (short)0, c,
                                                 false, false);
}

// ---------------------------------------------------------------------------
// Build concat inputs: [emb[id], feats] -> bf16 [N, 192]
// ---------------------------------------------------------------------------
__global__ void build_x(const float* __restrict__ x, const float* __restrict__ emb,
                        bf16* __restrict__ out, int N) {
  int t = blockIdx.x * blockDim.x + threadIdx.x;
  int total = N * 192;
  if (t >= total) return;
  int n = t / 192, c = t % 192;
  float v;
  if (c < 64) {
    int id = (int)x[(size_t)n * 129];
    v = emb[(size_t)id * 64 + c];
  } else {
    v = x[(size_t)n * 129 + 1 + (c - 64)];
  }
  out[t] = (bf16)v;
}

// W [K,N] f32 row-major -> Wt [N,K] bf16 (WMMA B-operand friendly)
__global__ void wconv(const float* __restrict__ W, bf16* __restrict__ Wt,
                      int K, int N) {
  int t = blockIdx.x * blockDim.x + threadIdx.x;
  if (t >= K * N) return;
  int n = t / K, k = t % K;
  Wt[t] = (bf16)W[(size_t)k * N + n];
}

__global__ void f32_to_bf16(const float* __restrict__ in, bf16* __restrict__ out,
                            int n) {
  int t = blockIdx.x * blockDim.x + threadIdx.x;
  if (t < n) out[t] = (bf16)in[t];
}

// ---------------------------------------------------------------------------
// Generic 16x16-tile WMMA GEMM:  C[M,N] = A[M,K] @ Wt[N,K]^T + bias
// One wave per C tile. K multiple of 32, M,N multiples of 16.
// mode 0: f32 C     mode 1: bf16 Cb (with colOff into a concat buffer)
// mode 2: bf16 head-major out[h][row][d]  (for Q/K packing, col = h*64+d)
// ---------------------------------------------------------------------------
__global__ __launch_bounds__(32)
void gemm16(const bf16* __restrict__ A, int lda,
            const bf16* __restrict__ Wt, int K,
            const float* __restrict__ bias,
            float* __restrict__ C, int ldc,
            bf16* __restrict__ Cb, int ldcb, int colOff,
            int mode, int strideH) {
  const int lane = threadIdx.x;
  const int half = lane >> 4;     // hi/lo 16 lanes
  const int l16  = lane & 15;
  const int rowbase = blockIdx.y * 16;
  const int colbase = blockIdx.x * 16;

  const bf16* arow = A  + (size_t)(rowbase + l16) * lda;
  const bf16* brow = Wt + (size_t)(colbase + l16) * K;

  v8f acc = {};
  for (int k0 = 0; k0 < K; k0 += 32) {
    // A 16-bit layout: lanes 0-15 hold K {0..7,16..23}, lanes 16-31 {8..15,24..31}
    int aoff = k0 + (half ? 8 : 0);
    v8bf a_lo = *(const v8bf*)(arow + aoff);
    v8bf a_hi = *(const v8bf*)(arow + aoff + 16);
    // B 16-bit layout: lanes 0-15 hold K {0..7,16..23}, lanes 16-31 {8..15,24..31} (N = l16)
    int boff = k0 + half * 8;
    v8bf b_lo = *(const v8bf*)(brow + boff);
    v8bf b_hi = *(const v8bf*)(brow + boff + 16);
    acc = wmma_bf16(cat16(a_lo, a_hi), cat16(b_lo, b_hi), acc);
  }

  float bb = bias ? bias[colbase + l16] : 0.f;
  if (mode == 2) {
    int col = colbase + l16;
    int h = col >> 6, d = col & 63;
    bf16* outh = Cb + (size_t)h * strideH;
#pragma unroll
    for (int r = 0; r < 8; ++r) {
      int grow = rowbase + r + half * 8;
      outh[(size_t)grow * 64 + d] = (bf16)(acc[r] + bb);
    }
  } else if (mode == 1) {
    int cidx = colOff + colbase + l16;
#pragma unroll
    for (int r = 0; r < 8; ++r) {
      int grow = rowbase + r + half * 8;
      Cb[(size_t)grow * ldcb + cidx] = (bf16)(acc[r] + bb);
    }
  } else {
    int cidx = colOff + colbase + l16;
#pragma unroll
    for (int r = 0; r < 8; ++r) {
      int grow = rowbase + r + half * 8;
      C[(size_t)grow * ldc + cidx] = acc[r] + bb;
    }
  }
}

// ---------------------------------------------------------------------------
// Fused attention: per (head, 16-row tile) block (128 thr = 4 waves).
// Each wave WMMAs 16x16 score tiles over its quarter of the 512 column
// tiles, applies mask*logit -> sigmoid, streams a per-half-row top-10.
// Lists merged through LDS; softmax over top-10, thresholded at 0.5.
// Emits 10 (idx, weight) pairs per (head, rna-row).
// ---------------------------------------------------------------------------
__global__ __launch_bounds__(128)
void attn_topk(const bf16* __restrict__ Qbf,   // [H][NR][64]
               const bf16* __restrict__ Kbf,   // [H][NA][64]
               const float* __restrict__ mask, // [NR][NA]
               int* __restrict__ topi, float* __restrict__ topw) {
  __shared__ float tile[4][256];
  __shared__ float lv[128 * TOPK];
  __shared__ int   li[128 * TOPK];

  const int tid  = threadIdx.x;
  const int wave = tid >> 5;
  const int lane = tid & 31;
  const int half = lane >> 4;
  const int l16  = lane & 15;
  const int h       = blockIdx.x >> 7;          // NR/16 = 128 row tiles
  const int rowbase = (blockIdx.x & 127) * 16;

  // Q A-operand (K=64 -> two k-steps), identical across the 4 waves (L0 hit)
  const bf16* qrow = Qbf + ((size_t)h * NR + rowbase + l16) * 64;
  int asel = half ? 8 : 0;
  v16bf A0 = cat16(*(const v8bf*)(qrow + asel),      *(const v8bf*)(qrow + asel + 16));
  v16bf A1 = cat16(*(const v8bf*)(qrow + 32 + asel), *(const v8bf*)(qrow + 32 + asel + 16));

  float vals[TOPK]; int idxs[TOPK];
#pragma unroll
  for (int i = 0; i < TOPK; ++i) { vals[i] = -1.f; idxs[i] = 0; }
  float vmin = -1.f; int imin = 0;

  const int myrow = l16;
  const int cbase = half * 8;

  const int mt0 = wave * 128;
  for (int mi = 0; mi < 128; ++mi) {
    const int mt = mt0 + mi;
    const bf16* krow = Kbf + ((size_t)h * NA + mt * 16 + l16) * 64;
    int bsel = half * 8;
    v16bf B0 = cat16(*(const v8bf*)(krow + bsel),      *(const v8bf*)(krow + bsel + 16));
    v16bf B1 = cat16(*(const v8bf*)(krow + 32 + bsel), *(const v8bf*)(krow + 32 + bsel + 16));

    v8f acc = {};
    acc = wmma_bf16(A0, B0, acc);
    acc = wmma_bf16(A1, B1, acc);

    // mask * logit -> sigmoid -> LDS tile (per-wave buffer)
#pragma unroll
    for (int r = 0; r < 8; ++r) {
      int grow = rowbase + r + half * 8;
      int col  = mt * 16 + l16;
      float m = mask[(size_t)grow * NA + col];
      float x = acc[r] * m;
      float s = 1.f / (1.f + __expf(-x));
      tile[wave][(r + half * 8) * 16 + l16] = s;
    }
    // single-wave LDS exchange: DS ops are in-order per wave; wait for data
    asm volatile("s_wait_dscnt 0" ::: "memory");

    // streaming top-10 over my 8 columns of my row
#pragma unroll
    for (int j = 0; j < 8; ++j) {
      float v = tile[wave][myrow * 16 + cbase + j];
      if (v > vmin) {
        int c = mt * 16 + cbase + j;
#pragma unroll
        for (int i = 0; i < TOPK; ++i) if (i == imin) { vals[i] = v; idxs[i] = c; }
        vmin = vals[0]; imin = 0;
#pragma unroll
        for (int i = 1; i < TOPK; ++i) if (vals[i] < vmin) { vmin = vals[i]; imin = i; }
      }
    }
    asm volatile("s_wait_dscnt 0" ::: "memory");
  }

  // publish candidate lists
#pragma unroll
  for (int i = 0; i < TOPK; ++i) { lv[tid * TOPK + i] = vals[i]; li[tid * TOPK + i] = idxs[i]; }
  __syncthreads();

  // rows 0..15: merge 8 lists (4 waves x 2 halves) of 10 -> global top-10
  if (tid < 16) {
    const int row = tid;
    float tv[TOPK]; int ti[TOPK];
#pragma unroll
    for (int k = 0; k < TOPK; ++k) {
      float bv = -3.f; int boff = 0;
      for (int s = 0; s < 8; ++s) {
        int lb = ((s >> 1) * 32 + (s & 1) * 16 + row) * TOPK;
        for (int q = 0; q < TOPK; ++q) {
          float v = lv[lb + q];
          if (v > bv) { bv = v; boff = lb + q; }
        }
      }
      tv[k] = bv; ti[k] = li[boff]; lv[boff] = -3.f;
    }
    // softmax over the 10 raw sigmoid scores, then threshold (>0.5)
    float mx = tv[0];
    float e[TOPK]; float sum = 0.f;
#pragma unroll
    for (int k = 0; k < TOPK; ++k) { e[k] = __expf(tv[k] - mx); sum += e[k]; }
    float inv = 1.f / sum;
    size_t base = ((size_t)h * NR + rowbase + row) * TOPK;
#pragma unroll
    for (int k = 0; k < TOPK; ++k) {
      float w = e[k] * inv * (tv[k] > 0.5f ? 1.f : 0.f);
      topw[base + k] = w;
      topi[base + k] = ti[k];
    }
  }
}

// r2a[n, h*64+d] = sum_k w * va[idx_k, h*64+d]  (bf16 out feeds opr GEMM)
__global__ void gather_r2a(const float* __restrict__ va32,
                           const int* __restrict__ topi,
                           const float* __restrict__ topw,
                           bf16* __restrict__ r2a_bf) {
  int t = blockIdx.x * blockDim.x + threadIdx.x;
  if (t >= NR * 256) return;
  int n = t >> 8, hd = t & 255, h = hd >> 6;
  size_t base = ((size_t)h * NR + n) * TOPK;
  float acc = 0.f;
#pragma unroll
  for (int k = 0; k < TOPK; ++k) {
    float w = topw[base + k];
    acc += w * va32[(size_t)topi[base + k] * 256 + hd];
  }
  r2a_bf[t] = (bf16)acc;
}

// a2r[idx_k, h*64+d] += w * vr[n, h*64+d]
__global__ void scatter_a2r(const float* __restrict__ vr32,
                            const int* __restrict__ topi,
                            const float* __restrict__ topw,
                            float* __restrict__ a2r32) {
  int t = blockIdx.x * blockDim.x + threadIdx.x;
  if (t >= NR * 256) return;
  int n = t >> 8, hd = t & 255, h = hd >> 6;
  float v = vr32[t];
  size_t base = ((size_t)h * NR + n) * TOPK;
#pragma unroll
  for (int k = 0; k < TOPK; ++k) {
    float w = topw[base + k];
    if (w != 0.f)
      atomicAdd(&a2r32[(size_t)topi[base + k] * 256 + hd], w * v);
  }
}

__global__ void segsum(const float* __restrict__ x, const int* __restrict__ batch,
                       float* __restrict__ sum, float* __restrict__ cnt, int N) {
  int t = blockIdx.x * blockDim.x + threadIdx.x;
  if (t >= N * 256) return;
  int n = t >> 8, c = t & 255;
  int b = batch[n];
  atomicAdd(&sum[b * 256 + c], x[t]);
  if (c == 0) atomicAdd(&cnt[b], 1.f);
}

// Tiny head MLP, B=8 rows: means -> fc1(relu) -> fc2(relu) -> fc3
__global__ __launch_bounds__(256)
void head_mlp(const float* __restrict__ sumr, const float* __restrict__ cntr,
              const float* __restrict__ suma, const float* __restrict__ cnta,
              const float* __restrict__ w1, const float* __restrict__ b1,
              const float* __restrict__ w2, const float* __restrict__ b2,
              const float* __restrict__ w3, const float* __restrict__ b3,
              float* __restrict__ out) {
  __shared__ float xb[8 * 512];
  __shared__ float h1[8 * 256];
  __shared__ float h2[8 * 128];
  int t = threadIdx.x;
  for (int i = t; i < 8 * 256; i += 256) {
    int b = i >> 8, c = i & 255;
    xb[b * 512 + c]       = sumr[i] / fmaxf(cntr[b], 1.f);
    xb[b * 512 + 256 + c] = suma[i] / fmaxf(cnta[b], 1.f);
  }
  __syncthreads();
  {
    int col = t;
    for (int b = 0; b < 8; ++b) {
      float a = b1[col];
      for (int k = 0; k < 512; ++k) a += xb[b * 512 + k] * w1[(size_t)k * 256 + col];
      h1[b * 256 + col] = fmaxf(a, 0.f);
    }
  }
  __syncthreads();
  if (t < 128) {
    int col = t;
    for (int b = 0; b < 8; ++b) {
      float a = b2[col];
      for (int k = 0; k < 256; ++k) a += h1[b * 256 + k] * w2[(size_t)k * 128 + col];
      h2[b * 128 + col] = fmaxf(a, 0.f);
    }
  }
  __syncthreads();
  if (t < 16) {
    int b = t >> 1, o = t & 1;
    float a = b3[o];
    for (int k = 0; k < 128; ++k) a += h2[b * 128 + k] * w3[(size_t)k * 2 + o];
    out[b * 2 + o] = a;
  }
}

// ---------------------------------------------------------------------------
extern "C" void kernel_launch(void* const* d_in, const int* in_sizes, int n_in,
                              void* d_out, int out_size, void* d_ws, size_t ws_size,
                              hipStream_t stream) {
  (void)in_sizes; (void)n_in; (void)out_size; (void)ws_size;

  const float* x_rna   = (const float*)d_in[0];
  const float* x_atac  = (const float*)d_in[1];
  const float* chrom   = (const float*)d_in[2];
  const int*   rna_b   = (const int*)d_in[3];
  const int*   atac_b  = (const int*)d_in[4];
  const float* rna_emb  = (const float*)d_in[5];
  const float* atac_emb = (const float*)d_in[6];
  const float *qw = (const float*)d_in[7],  *qb = (const float*)d_in[8];
  const float *kw = (const float*)d_in[9],  *kb = (const float*)d_in[10];
  const float *vrw = (const float*)d_in[11], *vrb = (const float*)d_in[12];
  const float *vaw = (const float*)d_in[13], *vab = (const float*)d_in[14];
  const float *oprw = (const float*)d_in[15], *oprb = (const float*)d_in[16];
  const float *opaw = (const float*)d_in[17], *opab = (const float*)d_in[18];
  const float *srw = (const float*)d_in[19], *srb = (const float*)d_in[20];
  const float *saw = (const float*)d_in[21], *sab = (const float*)d_in[22];
  const float *drw = (const float*)d_in[23], *drb = (const float*)d_in[24];
  const float *daw = (const float*)d_in[25], *dab = (const float*)d_in[26];
  const float *f1w = (const float*)d_in[27], *f1b = (const float*)d_in[28];
  const float *f2w = (const float*)d_in[29], *f2b = (const float*)d_in[30];
  const float *f3w = (const float*)d_in[31], *f3b = (const float*)d_in[32];

  size_t off = 0;
  auto alloc = [&](size_t bytes) -> char* {
    char* r = (char*)d_ws + off;
    off += (bytes + 255) & ~(size_t)255;
    return r;
  };
  bf16* wtq   = (bf16*)alloc(192 * 256 * 2);
  bf16* wtk   = (bf16*)alloc(192 * 256 * 2);
  bf16* wtvr  = (bf16*)alloc(192 * 256 * 2);
  bf16* wtva  = (bf16*)alloc(192 * 256 * 2);
  bf16* wtsr  = (bf16*)alloc(192 * 256 * 2);
  bf16* wtsa  = (bf16*)alloc(192 * 256 * 2);
  bf16* wtopr = (bf16*)alloc(256 * 256 * 2);
  bf16* wtopa = (bf16*)alloc(256 * 256 * 2);
  bf16* wtdr  = (bf16*)alloc(512 * 256 * 2);
  bf16* wtda  = (bf16*)alloc(512 * 256 * 2);
  bf16* xrbf  = (bf16*)alloc((size_t)NR * 192 * 2);
  bf16* xabf  = (bf16*)alloc((size_t)NA * 192 * 2);
  bf16* Qbf   = (bf16*)alloc((size_t)NHEAD * NR * 64 * 2);
  bf16* Kbf   = (bf16*)alloc((size_t)NHEAD * NA * 64 * 2);
  float* vr32 = (float*)alloc((size_t)NR * 256 * 4);
  float* va32 = (float*)alloc((size_t)NA * 256 * 4);
  int*   topi = (int*)alloc((size_t)NHEAD * NR * TOPK * 4);
  float* topw = (float*)alloc((size_t)NHEAD * NR * TOPK * 4);
  bf16* r2abf = (bf16*)alloc((size_t)NR * 256 * 2);
  float* a2r32 = (float*)alloc((size_t)NA * 256 * 4);
  bf16* a2rbf = (bf16*)alloc((size_t)NA * 256 * 2);
  bf16* drcat = (bf16*)alloc((size_t)NR * 512 * 2);
  bf16* dacat = (bf16*)alloc((size_t)NA * 512 * 2);
  float* xrh  = (float*)alloc((size_t)NR * 256 * 4);
  float* xah  = (float*)alloc((size_t)NA * 256 * 4);
  float* sums = (float*)alloc((8 * 256 * 2 + 16) * 4);
  float* sumr = sums;
  float* suma = sums + 8 * 256;
  float* cntr = sums + 8 * 256 * 2;
  float* cnta = cntr + 8;

  hipMemsetAsync(a2r32, 0, (size_t)NA * 256 * 4, stream);
  hipMemsetAsync(sums, 0, (8 * 256 * 2 + 16) * 4, stream);

  // weight convert+transpose
  auto WC = [&](const float* W, bf16* Wt, int K, int N) {
    int n = K * N;
    wconv<<<(n + 255) / 256, 256, 0, stream>>>(W, Wt, K, N);
  };
  WC(qw, wtq, 192, 256);  WC(kw, wtk, 192, 256);
  WC(vrw, wtvr, 192, 256); WC(vaw, wtva, 192, 256);
  WC(srw, wtsr, 192, 256); WC(saw, wtsa, 192, 256);
  WC(oprw, wtopr, 256, 256); WC(opaw, wtopa, 256, 256);
  WC(drw, wtdr, 512, 256);  WC(daw, wtda, 512, 256);

  build_x<<<(NR * 192 + 255) / 256, 256, 0, stream>>>(x_rna, rna_emb, xrbf, NR);
  build_x<<<(NA * 192 + 255) / 256, 256, 0, stream>>>(x_atac, atac_emb, xabf, NA);

  // projections (WMMA)
  gemm16<<<dim3(16, NR / 16), 32, 0, stream>>>(xrbf, 192, wtq, 192, qb,
      nullptr, 0, Qbf, 0, 0, 2, NR * 64);
  gemm16<<<dim3(16, NA / 16), 32, 0, stream>>>(xabf, 192, wtk, 192, kb,
      nullptr, 0, Kbf, 0, 0, 2, NA * 64);
  gemm16<<<dim3(16, NR / 16), 32, 0, stream>>>(xrbf, 192, wtvr, 192, vrb,
      vr32, 256, nullptr, 0, 0, 0, 0);
  gemm16<<<dim3(16, NA / 16), 32, 0, stream>>>(xabf, 192, wtva, 192, vab,
      va32, 256, nullptr, 0, 0, 0, 0);
  gemm16<<<dim3(16, NR / 16), 32, 0, stream>>>(xrbf, 192, wtsr, 192, srb,
      nullptr, 0, drcat, 512, 256, 1, 0);
  gemm16<<<dim3(16, NA / 16), 32, 0, stream>>>(xabf, 192, wtsa, 192, sab,
      nullptr, 0, dacat, 512, 256, 1, 0);

  // fused attention + top-k
  attn_topk<<<NHEAD * (NR / 16), 128, 0, stream>>>(Qbf, Kbf, chrom, topi, topw);

  // sparse aggregation
  gather_r2a<<<(NR * 256 + 255) / 256, 256, 0, stream>>>(va32, topi, topw, r2abf);
  scatter_a2r<<<(NR * 256 + 255) / 256, 256, 0, stream>>>(vr32, topi, topw, a2r32);
  f32_to_bf16<<<(NA * 256 + 255) / 256, 256, 0, stream>>>(a2r32, a2rbf, NA * 256);

  // post GEMMs into concat buffers, then fused dense layer
  gemm16<<<dim3(16, NR / 16), 32, 0, stream>>>(r2abf, 256, wtopr, 256, oprb,
      nullptr, 0, drcat, 512, 0, 1, 0);
  gemm16<<<dim3(16, NA / 16), 32, 0, stream>>>(a2rbf, 256, wtopa, 256, opab,
      nullptr, 0, dacat, 512, 0, 1, 0);
  gemm16<<<dim3(16, NR / 16), 32, 0, stream>>>(drcat, 512, wtdr, 512, drb,
      xrh, 256, nullptr, 0, 0, 0, 0);
  gemm16<<<dim3(16, NA / 16), 32, 0, stream>>>(dacat, 512, wtda, 512, dab,
      xah, 256, nullptr, 0, 0, 0, 0);

  // segment means + head MLP
  segsum<<<(NR * 256 + 255) / 256, 256, 0, stream>>>(xrh, rna_b, sumr, cntr, NR);
  segsum<<<(NA * 256 + 255) / 256, 256, 0, stream>>>(xah, atac_b, suma, cnta, NA);
  head_mlp<<<1, 256, 0, stream>>>(sumr, cntr, suma, cnta,
                                  f1w, f1b, f2w, f2b, f3w, f3b, (float*)d_out);
}
